// EEGMPNN_51711406243990
// MI455X (gfx1250) — matrix operand order repaired and checked by reading, not compiled, not run on hardware
//
#include <hip/hip_runtime.h>

typedef __attribute__((ext_vector_type(16))) __bf16 v16bf;
typedef __attribute__((ext_vector_type(8)))  float  v8f;

union FragU { v16bf v; unsigned u[8]; };

// ---------------------------------------------------------------------------
// Fragment loaders per CDNA5 WMMA 16-bit layouts (CDNA5 ISA §7.12.2)
// A (16x32, MxK): lanes 0-15 -> M=lane, K pairs {0..7,16..23}; lanes 16-31 ->
//   same M, K pairs shifted by 8. All K offsets even -> 32-bit LDS loads.
// B (32x16, KxN): lane -> N=lane&15; same K pairs as A ({0..7,16..23} for
//   lanes<16, shifted by 8 for lanes>=16)  (B[k][n] == W[n][k] since out = act @ W^T).
// ---------------------------------------------------------------------------
__device__ inline v16bf lds_load_A(const __bf16* t, int stride, int lane, int koff) {
  const int M  = lane & 15;
  const int kb = (lane & 16) ? 8 : 0;
  FragU a;
#pragma unroll
  for (int r = 0; r < 8; ++r) {
    int k = koff + ((r < 4) ? (kb + 2 * r) : (16 + kb + 2 * (r - 4)));
    a.u[r] = *(const unsigned*)(t + M * stride + k);
  }
  return a.v;
}

__device__ inline v16bf gbl_load_B(const __bf16* w, int ldw, int lane, int n0, int koff) {
  const int n  = n0 + (lane & 15);
  const int kb = (lane & 16) ? 8 : 0;
  FragU b;
#pragma unroll
  for (int r = 0; r < 8; ++r) {
    int k = koff + ((r < 4) ? (kb + 2 * r) : (16 + kb + 2 * (r - 4)));
    b.u[r] = *(const unsigned*)(w + n * ldw + k);
  }
  return b.v;
}

__device__ inline v8f wmma_bf16(v16bf a, v16bf b, v8f c) {
  return __builtin_amdgcn_wmma_f32_16x16x32_bf16(false, a, false, b, (short)0, c, false, false);
}

// f32 -> bf16 weight conversion (run fresh each launch; no cached state)
__global__ __launch_bounds__(256) void f2bf_kernel(const float* __restrict__ s,
                                                   __bf16* __restrict__ d, int n) {
  int i = blockIdx.x * 256 + threadIdx.x;
  if (i < n) d[i] = (__bf16)s[i];
}

// ---------------------------------------------------------------------------
// Edge kernel (persistent waves, grid-stride over 16-edge tiles).
// Weight B-fragments + bias/gate vectors are wave-invariant: hoisted into
// registers once; inner loop has only gather, 16 WMMAs, and scatter atomics.
//   h = relu(x[src] @ m1_w^T + m1_b); m = h @ m2_w^T + m2_b;
//   m *= edge_attr * e_w;  atomicAdd into agg[tgt]
// ---------------------------------------------------------------------------
__global__ __launch_bounds__(256) void edge_msg_kernel(
    const float* __restrict__ xin, const int* __restrict__ src, const int* __restrict__ tgt,
    const float* __restrict__ eattr,
    const __bf16* __restrict__ w1, const float* __restrict__ b1,
    const __bf16* __restrict__ w2, const float* __restrict__ b2,
    const float* __restrict__ ew, float* __restrict__ agg, int E)
{
  __shared__ __bf16 ldsT[8][16 * 68];           // per-wave 16x64 tile, padded stride
  const int wave = threadIdx.x >> 5;
  const int lane = threadIdx.x & 31;
  const int ncol = lane & 15;
  const int mofs = (lane & 16) ? 8 : 0;
  __bf16* t = &ldsT[wave][0];

  // ---- hoisted wave-invariant state: 16 B fragments + per-column constants
  v16bf B1[2][4], B2[2][4];
#pragma unroll
  for (int ks = 0; ks < 2; ++ks)
#pragma unroll
    for (int nt = 0; nt < 4; ++nt) {
      B1[ks][nt] = gbl_load_B(w1, 64, lane, nt * 16, ks * 32);
      B2[ks][nt] = gbl_load_B(w2, 64, lane, nt * 16, ks * 32);
    }
  float bias1[4], bias2[4], gate[4];
#pragma unroll
  for (int nt = 0; nt < 4; ++nt) {
    bias1[nt] = b1[nt * 16 + ncol];
    bias2[nt] = b2[nt * 16 + ncol];
    gate[nt]  = ew[nt * 16 + ncol];
  }

  const int ntiles = E >> 4;
  const int row  = lane >> 1;          // cooperative-gather row for this lane
  const int colb = (lane & 1) * 32;    // and its 32-col half

  for (int tile = blockIdx.x * 8 + wave; tile < ntiles; tile += gridDim.x * 8) {
    const int ebase = tile * 16;

    { // cooperative gather of 16 source rows (f32 -> bf16 into LDS)
      const int node = src[ebase + row];
      const float* xr = xin + (size_t)node * 64 + colb;
#pragma unroll
      for (int i = 0; i < 32; i += 4) {
        float4 v = *(const float4*)(xr + i);
        __bf16* dp = t + row * 68 + colb + i;
        dp[0] = (__bf16)v.x; dp[1] = (__bf16)v.y; dp[2] = (__bf16)v.z; dp[3] = (__bf16)v.w;
      }
    }
    asm volatile("s_wait_dscnt 0x0" ::: "memory");

    // GEMM1: h = relu(x @ w1^T + b1)   (K=64: 2 k-steps x 4 n-tiles)
    v8f h[4] = {};
#pragma unroll
    for (int ks = 0; ks < 2; ++ks) {
      v16bf a = lds_load_A(t, 68, lane, ks * 32);
#pragma unroll
      for (int nt = 0; nt < 4; ++nt) h[nt] = wmma_bf16(a, B1[ks][nt], h[nt]);
    }
#pragma unroll
    for (int nt = 0; nt < 4; ++nt) {
#pragma unroll
      for (int j = 0; j < 8; ++j) {
        float v = fmaxf(h[nt][j] + bias1[nt], 0.0f);
        t[(j + mofs) * 68 + nt * 16 + ncol] = (__bf16)v;   // C-layout -> tile
      }
    }
    asm volatile("s_wait_dscnt 0x0" ::: "memory");

    // GEMM2: m = h @ w2^T + b2
    v8f m[4] = {};
#pragma unroll
    for (int ks = 0; ks < 2; ++ks) {
      v16bf a = lds_load_A(t, 68, lane, ks * 32);
#pragma unroll
      for (int nt = 0; nt < 4; ++nt) m[nt] = wmma_bf16(a, B2[ks][nt], m[nt]);
    }

    float attr[8]; int trow[8];
#pragma unroll
    for (int j = 0; j < 8; ++j) {
      int e = ebase + j + mofs;
      attr[j] = eattr[e];
      trow[j] = tgt[e];
    }
#pragma unroll
    for (int nt = 0; nt < 4; ++nt) {
      int o = nt * 16 + ncol;
#pragma unroll
      for (int j = 0; j < 8; ++j) {
        float v = (m[nt][j] + bias2[nt]) * attr[j] * gate[nt];
        atomicAdd(&agg[(size_t)trow[j] * 64 + o], v);      // global_atomic_add_f32
      }
    }
  }
}

// ---------------------------------------------------------------------------
// Node kernel: out = relu(concat([x,agg]) @ u1^T + u1_b) @ u2^T + u2_b
// (16x fewer tiles than edges; streamed weights are fine here)
// ---------------------------------------------------------------------------
__global__ __launch_bounds__(256) void node_update_kernel(
    const float* __restrict__ xin, const float* __restrict__ agg,
    const __bf16* __restrict__ w1, const float* __restrict__ b1,
    const __bf16* __restrict__ w2, const float* __restrict__ b2,
    float* __restrict__ xout, int N, int relu_out)
{
  __shared__ __bf16 ldsT[8][16 * 136];   // 16x128 bf16 tile (padded)
  __shared__ __bf16 ldsH[8][16 * 68];    // 16x64 hidden tile
  const int wave  = threadIdx.x >> 5;
  const int lane  = threadIdx.x & 31;
  const int nbase = (blockIdx.x * 8 + wave) * 16;
  if (nbase >= N) return;
  __bf16* t  = &ldsT[wave][0];
  __bf16* th = &ldsH[wave][0];

  { // cooperative load: cols 0..63 from x, 64..127 from agg (concat)
    const int row  = lane >> 1;
    const int half = lane & 1;
    const float* sp = (half ? agg : xin) + (size_t)(nbase + row) * 64;
#pragma unroll
    for (int i = 0; i < 64; i += 4) {
      float4 v = *(const float4*)(sp + i);
      __bf16* dp = t + row * 136 + half * 64 + i;
      dp[0] = (__bf16)v.x; dp[1] = (__bf16)v.y; dp[2] = (__bf16)v.z; dp[3] = (__bf16)v.w;
    }
  }
  asm volatile("s_wait_dscnt 0x0" ::: "memory");

  const int ncol = lane & 15;
  const int mofs = (lane & 16) ? 8 : 0;

  // GEMM1: K=128 -> 4 k-steps
  v8f h[4] = {};
#pragma unroll
  for (int ks = 0; ks < 4; ++ks) {
    v16bf a = lds_load_A(t, 136, lane, ks * 32);
#pragma unroll
    for (int nt = 0; nt < 4; ++nt) {
      v16bf b = gbl_load_B(w1, 128, lane, nt * 16, ks * 32);
      h[nt] = wmma_bf16(a, b, h[nt]);
    }
  }
#pragma unroll
  for (int nt = 0; nt < 4; ++nt) {
    float bias = b1[nt * 16 + ncol];
#pragma unroll
    for (int j = 0; j < 8; ++j) {
      float v = fmaxf(h[nt][j] + bias, 0.0f);
      th[(j + mofs) * 68 + nt * 16 + ncol] = (__bf16)v;
    }
  }
  asm volatile("s_wait_dscnt 0x0" ::: "memory");

  // GEMM2: K=64
  v8f o[4] = {};
#pragma unroll
  for (int ks = 0; ks < 2; ++ks) {
    v16bf a = lds_load_A(th, 68, lane, ks * 32);
#pragma unroll
    for (int nt = 0; nt < 4; ++nt) {
      v16bf b = gbl_load_B(w2, 64, lane, nt * 16, ks * 32);
      o[nt] = wmma_bf16(a, b, o[nt]);
    }
  }
#pragma unroll
  for (int nt = 0; nt < 4; ++nt) {
    float bias = b2[nt * 16 + ncol];
#pragma unroll
    for (int j = 0; j < 8; ++j) {
      float v = o[nt][j] + bias;
      if (relu_out) v = fmaxf(v, 0.0f);
      xout[(size_t)(nbase + j + mofs) * 64 + nt * 16 + ncol] = v;
    }
  }
}

// ---------------------------------------------------------------------------
// Mean pool + final linear
// ---------------------------------------------------------------------------
__global__ __launch_bounds__(256) void pool_accum_kernel(
    const float* __restrict__ x2, const int* __restrict__ batch,
    float* __restrict__ sums, float* __restrict__ cnt, int N)
{
  long long idx = (long long)blockIdx.x * 256 + threadIdx.x;
  if (idx >= (long long)N * 64) return;
  int n = (int)(idx >> 6), d = (int)(idx & 63);
  int g = batch[n];
  atomicAdd(&sums[(size_t)g * 64 + d], x2[idx]);
  if (d == 0) atomicAdd(&cnt[g], 1.0f);
}

__global__ __launch_bounds__(256) void final_lin_kernel(
    const float* __restrict__ sums, const float* __restrict__ cnt,
    const float* __restrict__ lw, const float* __restrict__ lb,
    float* __restrict__ out, int G)
{
  int g = blockIdx.x * 256 + threadIdx.x;
  if (g >= G) return;
  float acc = 0.0f;
#pragma unroll
  for (int d = 0; d < 64; ++d) acc += sums[(size_t)g * 64 + d] * lw[d];
  out[g] = acc / fmaxf(cnt[g], 1.0f) + lb[0];
}

// ---------------------------------------------------------------------------
// Host launch
// ---------------------------------------------------------------------------
struct LayerP { const float *ew, *m1w, *m1b, *m2w, *m2b, *u1w, *u1b, *u2w, *u2b; };

static LayerP map_layer(void* const* d_in, const int* sz, int base) {
  LayerP p;
  auto f = [&](int i) { return (const float*)d_in[i]; };
  if (sz[base] == 64 && sz[base + 2] == 4096) {
    // pytree alphabetical: e_w, m1_b, m1_w, m2_b, m2_w, u1_b, u1_w, u2_b, u2_w
    p.ew  = f(base + 0); p.m1b = f(base + 1); p.m1w = f(base + 2);
    p.m2b = f(base + 3); p.m2w = f(base + 4);
    p.u1b = f(base + 5); p.u1w = f(base + 6);
    p.u2b = f(base + 7); p.u2w = f(base + 8);
  } else {
    // insertion order: m1_w, m1_b, m2_w, m2_b, u1_w, u1_b, u2_w, u2_b, e_w
    p.m1w = f(base + 0); p.m1b = f(base + 1);
    p.m2w = f(base + 2); p.m2b = f(base + 3);
    p.u1w = f(base + 4); p.u1b = f(base + 5);
    p.u2w = f(base + 6); p.u2b = f(base + 7);
    p.ew  = f(base + 8);
  }
  return p;
}

extern "C" void kernel_launch(void* const* d_in, const int* in_sizes, int n_in,
                              void* d_out, int out_size, void* d_ws, size_t ws_size,
                              hipStream_t stream)
{
  const float* x     = (const float*)d_in[0];
  const int*   ei    = (const int*)d_in[1];   // [2][E]: src, tgt
  const int*   batch = (const int*)d_in[2];
  const float* eattr = (const float*)d_in[3];

  LayerP L0 = map_layer(d_in, in_sizes, 4);
  LayerP L1 = map_layer(d_in, in_sizes, 13);
  const float* lin_b;
  const float* lin_w;
  if (in_sizes[22] == 1) { lin_b = (const float*)d_in[22]; lin_w = (const float*)d_in[23]; }
  else                   { lin_w = (const float*)d_in[22]; lin_b = (const float*)d_in[23]; }

  const int N = in_sizes[2];
  const int E = in_sizes[3];
  const int G = out_size;  // 256

  // workspace carve
  float* agg  = (float*)d_ws;
  float* x1   = agg  + (size_t)N * 64;
  float* x2   = x1   + (size_t)N * 64;
  float* sums = x2   + (size_t)N * 64;
  float* cnt  = sums + (size_t)G * 64;
  __bf16* wb  = (__bf16*)(cnt + G);
  __bf16* b_m1_0 = wb;               __bf16* b_m2_0 = b_m1_0 + 4096;
  __bf16* b_u1_0 = b_m2_0 + 4096;    __bf16* b_u2_0 = b_u1_0 + 8192;
  __bf16* b_m1_1 = b_u2_0 + 4096;    __bf16* b_m2_1 = b_m1_1 + 4096;
  __bf16* b_u1_1 = b_m2_1 + 4096;    __bf16* b_u2_1 = b_u1_1 + 8192;

  // convert weights to bf16 (cheap, deterministic each call)
  f2bf_kernel<<<16, 256, 0, stream>>>(L0.m1w, b_m1_0, 4096);
  f2bf_kernel<<<16, 256, 0, stream>>>(L0.m2w, b_m2_0, 4096);
  f2bf_kernel<<<32, 256, 0, stream>>>(L0.u1w, b_u1_0, 8192);
  f2bf_kernel<<<16, 256, 0, stream>>>(L0.u2w, b_u2_0, 4096);
  f2bf_kernel<<<16, 256, 0, stream>>>(L1.m1w, b_m1_1, 4096);
  f2bf_kernel<<<16, 256, 0, stream>>>(L1.m2w, b_m2_1, 4096);
  f2bf_kernel<<<32, 256, 0, stream>>>(L1.u1w, b_u1_1, 8192);
  f2bf_kernel<<<16, 256, 0, stream>>>(L1.u2w, b_u2_1, 4096);

  hipMemsetAsync(agg, 0, (size_t)N * 64 * sizeof(float), stream);
  hipMemsetAsync(sums, 0, (size_t)(G * 64 + G) * sizeof(float), stream);

  // persistent-wave grid for the edge kernel (grid-stride over tiles)
  const int etiles  = E >> 4;
  int eblocks = 1250;                        // 10k persistent waves
  if (eblocks * 8 > etiles) eblocks = (etiles + 7) / 8;
  if (eblocks < 1) eblocks = 1;

  const int ntiles  = (N + 15) / 16;
  const int nblocks = (ntiles + 7) / 8;

  // layer 0 (output relu'd by node kernel)
  edge_msg_kernel<<<eblocks, 256, 0, stream>>>(x, ei, ei + E, eattr,
      b_m1_0, L0.m1b, b_m2_0, L0.m2b, L0.ew, agg, E);
  node_update_kernel<<<nblocks, 256, 0, stream>>>(x, agg,
      b_u1_0, L0.u1b, b_u2_0, L0.u2b, x1, N, /*relu_out=*/1);

  // layer 1
  hipMemsetAsync(agg, 0, (size_t)N * 64 * sizeof(float), stream);
  edge_msg_kernel<<<eblocks, 256, 0, stream>>>(x1, ei, ei + E, eattr,
      b_m1_1, L1.m1b, b_m2_1, L1.m2b, L1.ew, agg, E);
  node_update_kernel<<<nblocks, 256, 0, stream>>>(x1, agg,
      b_u1_1, L1.u1b, b_u2_1, L1.u2b, x2, N, /*relu_out=*/0);

  // global mean pool + final linear
  long long pelems = (long long)N * 64;
  pool_accum_kernel<<<(unsigned)((pelems + 255) / 256), 256, 0, stream>>>(x2, batch, sums, cnt, N);
  final_lin_kernel<<<(G + 255) / 256, 256, 0, stream>>>(sums, cnt, lin_w, lin_b, (float*)d_out, G);
}
